// SymplecticIntegrator_72593537237649
// MI455X (gfx1250) — matrix operand order repaired and checked by reading, not compiled, not run on hardware
//
#include <hip/hip_runtime.h>

// Problem constants (fixed by setup_inputs)
#define B_ 4096
#define D_ 1024
#define R_ 256
#define STEPS_ 8

typedef __bf16 bf16;
typedef bf16  bf16x8  __attribute__((ext_vector_type(8)));
typedef bf16  bf16x16 __attribute__((ext_vector_type(16)));
typedef float f32x8   __attribute__((ext_vector_type(8)));
typedef int   v4i     __attribute__((ext_vector_type(4)));

union Frag { bf16x16 v; bf16x8 h[2]; };

// ---------------------------------------------------------------------------
// CDNA5 async global->LDS copy (16B per lane), tracked with ASYNCcnt.
// Builtin prototype (probe-discovered): (v4i32 as1*, v4i32 as3*, imm, imm).
// ---------------------------------------------------------------------------
__device__ __forceinline__ void async_cp16(bf16* lds_dst, const bf16* gsrc) {
#if __has_builtin(__builtin_amdgcn_global_load_async_to_lds_b128)
    __builtin_amdgcn_global_load_async_to_lds_b128(
        (__attribute__((address_space(1))) v4i*)gsrc,
        (__attribute__((address_space(3))) v4i*)lds_dst, 0, 0);
#else
    unsigned laddr = (unsigned)(size_t)(__attribute__((address_space(3))) bf16*)lds_dst;
    asm volatile("global_load_async_to_lds_b128 %0, %1, off"
                 :: "v"(laddr), "v"(gsrc) : "memory");
#endif
}

__device__ __forceinline__ void wait_async0() {
#if __has_builtin(__builtin_amdgcn_s_wait_asynccnt)
    __builtin_amdgcn_s_wait_asynccnt(0);
#else
    asm volatile("s_wait_asynccnt 0" ::: "memory");
#endif
}

// ---------------------------------------------------------------------------
// Pack U (DxR) and W (RxD) into bf16 fragment-friendly layout:
// record idx = k16*N + n holds 16 contiguous bf16: src[(k16*16+e)*N + n].
// A lane loading B-fragment (kBase,n) reads record (kBase/16 + laneHalf)*N + n.
// ---------------------------------------------------------------------------
__global__ void pack_kernel(const float* __restrict__ U, const float* __restrict__ W,
                            bf16* __restrict__ Upack, bf16* __restrict__ Wpack) {
    int idx = blockIdx.x * blockDim.x + threadIdx.x;
    const int nU = (D_ / 16) * R_;   // 16384 records
    const int nW = (R_ / 16) * D_;   // 16384 records
    if (idx < nU) {
        int k16 = idx / R_;
        int n   = idx % R_;
        bf16* dst = Upack + (size_t)idx * 16;
        #pragma unroll
        for (int e = 0; e < 16; ++e)
            dst[e] = (bf16)U[(size_t)(k16 * 16 + e) * R_ + n];
    }
    if (idx < nW) {
        int k16 = idx / D_;
        int n   = idx % D_;
        bf16* dst = Wpack + (size_t)idx * 16;
        #pragma unroll
        for (int e = 0; e < 16; ++e)
            dst[e] = (bf16)W[(size_t)(k16 * 16 + e) * D_ + n];
    }
}

// Copy initial f32 state into d_out (live state) and create bf16 shadows.
__global__ void init_kernel(const float* __restrict__ x0, const float* __restrict__ v0,
                            float* __restrict__ xo, float* __restrict__ vo,
                            bf16* __restrict__ xbf, bf16* __restrict__ vbf) {
    size_t i = (size_t)blockIdx.x * blockDim.x + threadIdx.x;
    float xv = x0[i], vv = v0[i];
    xo[i] = xv;  vo[i] = vv;
    xbf[i] = (bf16)xv;  vbf[i] = (bf16)vv;
}

// LDS row padded to 40 bf16 (20 dwords): 16-lane ds_load_b128 fragment reads
// start at banks {0,4,...,60} -> conflict-free.
#define LROW 40

// ---------------------------------------------------------------------------
// chris_kernel: P[B,R] = tanh(x@U) * (v@U)^2  (bf16 in, f32 accumulate)
// Block tile 128(M) x 64(N); 8 waves as 4(M) x 2(N); wave tile 32x32,
// two accumulator sets (x@U and v@U) sharing the U B-fragments.
// A tiles (x,v) double-buffered in LDS via async copies; U direct from L2.
// ---------------------------------------------------------------------------
__launch_bounds__(256)
__global__ void chris_kernel(const bf16* __restrict__ xbf, const bf16* __restrict__ vbf,
                             const bf16* __restrict__ Upack, bf16* __restrict__ Pbf) {
    __shared__ __align__(16) bf16 ldsX[2][128][LROW];
    __shared__ __align__(16) bf16 ldsV[2][128][LROW];

    const int tid  = threadIdx.x;
    const int lane = tid & 31;
    const int wave = tid >> 5;
    const int mw   = wave & 3;       // 4 M subtiles of 32
    const int nw   = wave >> 2;      // 2 N subtiles of 32
    const int blockM = blockIdx.x * 128;
    const int tileN  = blockIdx.y * 64 + nw * 32;
    const int m    = lane & 15;
    const int half = lane >> 4;

    f32x8 accX[2][2] = {};
    f32x8 accV[2][2] = {};

    // prologue: async-stage chunk 0 into buffer 0
    {
        #pragma unroll
        for (int i = 0; i < 2; ++i) {
            int idx = i * 256 + tid;          // 0..511
            int row = idx >> 2, seg = idx & 3;
            size_t g = (size_t)(blockM + row) * D_ + seg * 8;
            async_cp16(&ldsX[0][row][seg * 8], xbf + g);
            async_cp16(&ldsV[0][row][seg * 8], vbf + g);
        }
    }

    const int NCHUNK = D_ / 32;
    for (int kc = 0; kc < NCHUNK; ++kc) {
        const int buf = kc & 1;
        const int k   = kc * 32;
        wait_async0();          // our copies for chunk kc have landed
        __syncthreads();        // everyone's copies landed; prior reads of buf^1 done

        if (kc + 1 < NCHUNK) {  // overlap copy of chunk kc+1 with compute of kc
            #pragma unroll
            for (int i = 0; i < 2; ++i) {
                int idx = i * 256 + tid;
                int row = idx >> 2, seg = idx & 3;
                size_t g = (size_t)(blockM + row) * D_ + (k + 32) + seg * 8;
                async_cp16(&ldsX[buf ^ 1][row][seg * 8], xbf + g);
                async_cp16(&ldsV[buf ^ 1][row][seg * 8], vbf + g);
            }
        }

        Frag ax[2], av[2], bu[2];
        #pragma unroll
        for (int mi = 0; mi < 2; ++mi) {
            // A-matrix 16-bit layout (contiguous, matches B): half 0 -> K {0..15}, half 1 -> {16..31}
            const int row = mw * 32 + mi * 16 + m;
            ax[mi].h[0] = *(const bf16x8*)&ldsX[buf][row][16 * half];
            ax[mi].h[1] = *(const bf16x8*)&ldsX[buf][row][16 * half + 8];
            av[mi].h[0] = *(const bf16x8*)&ldsV[buf][row][16 * half];
            av[mi].h[1] = *(const bf16x8*)&ldsV[buf][row][16 * half + 8];
        }
        #pragma unroll
        for (int ni = 0; ni < 2; ++ni) {
            // B-matrix: lane n = lane%16, half selects K+16; prepacked contiguous
            const bf16* bp = Upack + ((size_t)(k / 16 + half) * R_ + tileN + ni * 16 + m) * 16;
            bu[ni].h[0] = *(const bf16x8*)bp;
            bu[ni].h[1] = *(const bf16x8*)(bp + 8);
        }
        #pragma unroll
        for (int mi = 0; mi < 2; ++mi)
            #pragma unroll
            for (int ni = 0; ni < 2; ++ni) {
                accX[mi][ni] = __builtin_amdgcn_wmma_f32_16x16x32_bf16(
                    false, ax[mi].v, false, bu[ni].v, (short)0, accX[mi][ni], false, false);
                accV[mi][ni] = __builtin_amdgcn_wmma_f32_16x16x32_bf16(
                    false, av[mi].v, false, bu[ni].v, (short)0, accV[mi][ni], false, false);
            }
        __syncthreads();        // all reads of buf done before it is overwritten
    }

    // Elementwise: P = tanh(gx) * h * h ; C/D layout: n = lane%16, M = r + 8*half
    #pragma unroll
    for (int mi = 0; mi < 2; ++mi)
        #pragma unroll
        for (int ni = 0; ni < 2; ++ni)
            #pragma unroll
            for (int r = 0; r < 8; ++r) {
                int mg = blockM + mw * 32 + mi * 16 + r + 8 * half;
                int ng = tileN + ni * 16 + m;
                float g = tanhf(accX[mi][ni][r]);
                float h = accV[mi][ni][r];
                Pbf[(size_t)mg * R_ + ng] = (bf16)(g * h * h);
            }
}

// ---------------------------------------------------------------------------
// update_kernel: gamma = P@W fused with Verlet update.
// updateX=1: v = v + 0.5*dt*(force - gamma); x = x + dt*v  (half-kick + drift)
// updateX=0: v = v + 0.5*dt*(force - gamma)                (final half-kick)
// P tile double-buffered in LDS via async copies; W direct from L2.
// ---------------------------------------------------------------------------
__launch_bounds__(256)
__global__ void update_kernel(const bf16* __restrict__ Pbf, const bf16* __restrict__ Wpack,
                              const float* __restrict__ force,
                              float* __restrict__ x, float* __restrict__ v,
                              bf16* __restrict__ xbf, bf16* __restrict__ vbf,
                              int updateX) {
    __shared__ __align__(16) bf16 ldsP[2][128][LROW];

    const int tid  = threadIdx.x;
    const int lane = tid & 31;
    const int wave = tid >> 5;
    const int mw   = wave & 3;
    const int nw   = wave >> 2;
    const int blockM = blockIdx.x * 128;
    const int tileN  = blockIdx.y * 64 + nw * 32;   // over D_
    const int m    = lane & 15;
    const int half = lane >> 4;

    f32x8 acc[2][2] = {};

    // prologue: async-stage chunk 0
    {
        #pragma unroll
        for (int i = 0; i < 2; ++i) {
            int idx = i * 256 + tid;
            int row = idx >> 2, seg = idx & 3;
            async_cp16(&ldsP[0][row][seg * 8], Pbf + (size_t)(blockM + row) * R_ + seg * 8);
        }
    }

    const int NCHUNK = R_ / 32;
    for (int kc = 0; kc < NCHUNK; ++kc) {
        const int buf = kc & 1;
        const int k   = kc * 32;
        wait_async0();
        __syncthreads();

        if (kc + 1 < NCHUNK) {
            #pragma unroll
            for (int i = 0; i < 2; ++i) {
                int idx = i * 256 + tid;
                int row = idx >> 2, seg = idx & 3;
                async_cp16(&ldsP[buf ^ 1][row][seg * 8],
                           Pbf + (size_t)(blockM + row) * R_ + (k + 32) + seg * 8);
            }
        }

        Frag a[2], b[2];
        #pragma unroll
        for (int mi = 0; mi < 2; ++mi) {
            const int row = mw * 32 + mi * 16 + m;
            a[mi].h[0] = *(const bf16x8*)&ldsP[buf][row][16 * half];
            a[mi].h[1] = *(const bf16x8*)&ldsP[buf][row][16 * half + 8];
        }
        #pragma unroll
        for (int ni = 0; ni < 2; ++ni) {
            const bf16* bp = Wpack + ((size_t)(k / 16 + half) * D_ + tileN + ni * 16 + m) * 16;
            b[ni].h[0] = *(const bf16x8*)bp;
            b[ni].h[1] = *(const bf16x8*)(bp + 8);
        }
        #pragma unroll
        for (int mi = 0; mi < 2; ++mi)
            #pragma unroll
            for (int ni = 0; ni < 2; ++ni)
                acc[mi][ni] = __builtin_amdgcn_wmma_f32_16x16x32_bf16(
                    false, a[mi].v, false, b[ni].v, (short)0, acc[mi][ni], false, false);
        __syncthreads();
    }

    const float dt  = 0.01f;
    const float hdt = 0.5f * dt;
    #pragma unroll
    for (int mi = 0; mi < 2; ++mi)
        #pragma unroll
        for (int ni = 0; ni < 2; ++ni)
            #pragma unroll
            for (int r = 0; r < 8; ++r) {
                int mg = blockM + mw * 32 + mi * 16 + r + 8 * half;
                int ng = tileN + ni * 16 + m;
                size_t addr = (size_t)mg * D_ + ng;
                float gamma = acc[mi][ni][r];
                float vn = v[addr] + hdt * (force[addr] - gamma);
                v[addr]   = vn;
                vbf[addr] = (bf16)vn;
                if (updateX) {
                    float xn = x[addr] + dt * vn;
                    x[addr]   = xn;
                    xbf[addr] = (bf16)xn;
                }
            }
}

// ---------------------------------------------------------------------------
extern "C" void kernel_launch(void* const* d_in, const int* in_sizes, int n_in,
                              void* d_out, int out_size, void* d_ws, size_t ws_size,
                              hipStream_t stream) {
    const float* x0    = (const float*)d_in[0];
    const float* v0    = (const float*)d_in[1];
    const float* force = (const float*)d_in[2];
    const float* U     = (const float*)d_in[3];
    const float* W     = (const float*)d_in[4];
    // d_in[5] = steps (fixed at 8 by the harness setup)

    float* xo = (float*)d_out;                 // live x state (output region 0)
    float* vo = xo + (size_t)B_ * D_;          // live v state (output region 1)

    char* w = (char*)d_ws;
    bf16* Pbf   = (bf16*)w;                                   // B*R
    bf16* xbf   = (bf16*)(w + (size_t)B_ * R_ * 2);           // B*D
    bf16* vbf   = xbf + (size_t)B_ * D_;                      // B*D
    bf16* Upack = vbf + (size_t)B_ * D_;                      // D*R
    bf16* Wpack = Upack + (size_t)D_ * R_;                    // R*D
    // total ws usage: 2 + 8 + 8 + 0.5 + 0.5 = 19 MB

    pack_kernel<<<(16384 + 255) / 256, 256, 0, stream>>>(U, W, Upack, Wpack);
    init_kernel<<<(B_ * D_) / 256, 256, 0, stream>>>(x0, v0, xo, vo, xbf, vbf);

    dim3 gC(B_ / 128, R_ / 64);   // 32 x 4
    dim3 gU(B_ / 128, D_ / 64);   // 32 x 16
    for (int s = 0; s < STEPS_; ++s) {
        chris_kernel <<<gC, 256, 0, stream>>>(xbf, vbf, Upack, Pbf);
        update_kernel<<<gU, 256, 0, stream>>>(Pbf, Wpack, force, xo, vo, xbf, vbf, 1);
        chris_kernel <<<gC, 256, 0, stream>>>(xbf, vbf, Upack, Pbf);
        update_kernel<<<gU, 256, 0, stream>>>(Pbf, Wpack, force, xo, vo, xbf, vbf, 0);
    }
}
